// KurganovTadmorSchemeTorch_51127290692375
// MI455X (gfx1250) — matrix operand order repaired and checked by reading, not compiled, not run on hardware
//
#include <hip/hip_runtime.h>

typedef __attribute__((ext_vector_type(16))) _Float16 v16h;
typedef __attribute__((ext_vector_type(8)))  _Float16 h8;
typedef __attribute__((ext_vector_type(8)))  float    v8f;

#define NPOS  8192
#define NROWS 128
#define WAVES 8

// sigmoid via hardware v_exp_f32 + v_rcp_f32 (no IEEE-div expansion)
__device__ __forceinline__ float sigmoid_(float z) {
  return __builtin_amdgcn_rcpf(1.0f + __expf(-z));
}
// minmod(a,b) = sign(b) * max(0, min(|b|, sign(b)*a))
__device__ __forceinline__ float minmod_(float aa, float bb) {
  float sb = (bb > 0.0f) ? 1.0f : ((bb < 0.0f) ? -1.0f : 0.0f);
  return sb * fmaxf(0.0f, fminf(fabsf(bb), sb * aa));
}

// Computes H[row][e] for all N edges of each row, given state v (periodic).
// Each wave handles 32 edges = 4 MLP batches of 16 points (uL x2 tiles, uR x2 tiles).
__global__ __launch_bounds__(256) void kt_flux_kernel(
    const float* __restrict__ vin,
    const float* __restrict__ W1, const float* __restrict__ B1,
    const float* __restrict__ W2, const float* __restrict__ B2,
    const float* __restrict__ W3, const float* __restrict__ B3,
    float* __restrict__ Hout)
{
  __shared__ float    sW1[64];
  __shared__ float    sB1[64];
  __shared__ _Float16 sW2h[64 * 64];         // f16 copy of W2 (row-major): rows of W2 = cols of W2^T frags
  __shared__ float    sX[WAVES][64];         // per wave: [0:32]=uL, [32:64]=uR
  __shared__ _Float16 sG2[WAVES][16 * 64];   // backward-A staging (16 points x 64 feats)
  __shared__ float    sFD[WAVES][2][64];     // [0]=f, [1]=df per batch slot

  const int tid  = threadIdx.x;
  const int w    = tid >> 5;       // wave id
  const int l    = tid & 31;       // lane
  const int half = l >> 4;         // 0: lanes 0-15, 1: lanes 16-31
  const int n    = l & 15;         // column / row-within-16 index
  const int row  = blockIdx.y;

  // ---- stage shared weights ----
  if (tid < 64) { sW1[tid] = W1[tid]; sB1[tid] = B1[tid]; }
  for (int i = tid; i < 64 * 64; i += 256) sW2h[i] = (_Float16)W2[i];

  // ---- per-lane column constants (col k = nt*16 + n) ----
  float w1c[4], b1c[4], w3c[4], b2c[4];
#pragma unroll
  for (int nt = 0; nt < 4; ++nt) {
    w1c[nt] = W1[nt * 16 + n];
    b1c[nt] = B1[nt * 16 + n];
    w3c[nt] = W3[nt * 16 + n];
    b2c[nt] = B2[nt * 16 + n];
  }
  const float b3v = B3[0];

  // ---- forward W2 B-fragments (kept in registers) ----
  // B layout (16-bit, 32x16): lane holds col N=l%16, K = kt*32 + half*8 + {0..7} and +16..23
  v16h bw[2][4];
#pragma unroll
  for (int kt = 0; kt < 2; ++kt) {
#pragma unroll
    for (int nt = 0; nt < 4; ++nt) {
      const int k0 = kt * 32 + half * 8;
      const int nn = nt * 16 + n;
#pragma unroll
      for (int i = 0; i < 16; ++i)
        bw[kt][nt][i] = (_Float16)W2[(k0 + i + (i & 8)) * 64 + nn];
    }
  }

  // ---- phase 0: build uL/uR for this wave's 32 edges ----
  const int e0 = blockIdx.x * (WAVES * 32) + w * 32;
  const int e  = e0 + l;
  const float* vrow = vin + (size_t)row * NPOS;
  const float vm2 = vrow[(e - 2) & (NPOS - 1)];
  const float vm1 = vrow[(e - 1) & (NPOS - 1)];
  const float v0  = vrow[e];
  const float vp1 = vrow[(e + 1) & (NPOS - 1)];
  const float uLv = vm1 + 0.5f * minmod_(vm1 - vm2, v0 - vm1);   // cell e-1 right face
  const float uRv = v0  - 0.5f * minmod_(v0 - vm1, vp1 - v0);    // cell e left face
  sX[w][l]      = uLv;
  sX[w][32 + l] = uRv;
  __syncthreads();

  // ---- 4 batches of 16 MLP points each ----
#pragma unroll 1
  for (int bi = 0; bi < 4; ++bi) {
    const int side  = bi & 1;         // 0 = uL, 1 = uR
    const int t     = bi >> 1;        // tile within the wave
    const int xbase = side * 32 + t * 16;

    // x for this lane's A-fragment row (row = l%16)
    const float xr = sX[w][xbase + n];

    // -- phase A: layer 1 directly in A-fragment layout --
    // lane l holds row l%16; K = kt*32 + half*8 + {0..7} and +16..23
    v16h a[2];
#pragma unroll
    for (int kt = 0; kt < 2; ++kt) {
      const int kb = kt * 32 + half * 8;
#pragma unroll
      for (int i = 0; i < 8; ++i) {
        float za = fmaf(xr, sW1[kb + i],      sB1[kb + i]);
        float zb = fmaf(xr, sW1[kb + 16 + i], sB1[kb + 16 + i]);
        a[kt][i]     = (_Float16)(za * sigmoid_(za));
        a[kt][8 + i] = (_Float16)(zb * sigmoid_(zb));
      }
    }

    // -- phase B/C: forward h1@W2 (+b2), activation, f partials, g2 staging --
    float fp[8];
#pragma unroll
    for (int vv = 0; vv < 8; ++vv) fp[vv] = 0.0f;

#pragma unroll
    for (int nt = 0; nt < 4; ++nt) {
      v8f acc;
#pragma unroll
      for (int vv = 0; vv < 8; ++vv) acc[vv] = b2c[nt];   // bias: const per column
      acc = __builtin_amdgcn_wmma_f32_16x16x32_f16(false, a[0], false, bw[0][nt],
                                                   (short)0, acc, false, false);
      acc = __builtin_amdgcn_wmma_f32_16x16x32_f16(false, a[1], false, bw[1][nt],
                                                   (short)0, acc, false, false);
      // D layout: lane holds row M = vv + half*8, col N = nt*16 + n
#pragma unroll
      for (int vv = 0; vv < 8; ++vv) {
        float z  = acc[vv];
        float s  = sigmoid_(z);
        float h2 = z * s;
        float sp = s * (1.0f + z * (1.0f - s));   // silu'(z)
        fp[vv] = fmaf(h2, w3c[nt], fp[vv]);       // f partial: sum_k h2*W3[k]
        sG2[w][(vv + half * 8) * 64 + nt * 16 + n] = (_Float16)(w3c[nt] * sp);
      }
    }
    __syncthreads();

    // -- phase A': rebuild g2 in A-fragment layout from LDS --
    v16h ap[2];
#pragma unroll
    for (int kt = 0; kt < 2; ++kt) {
      const int kb = kt * 32 + half * 8;
      const _Float16* gr = &sG2[w][n * 64];
      h8 r0 = *(const h8*)(gr + kb);
      h8 r1 = *(const h8*)(gr + kb + 16);
#pragma unroll
      for (int i = 0; i < 8; ++i) { ap[kt][i] = r0[i]; ap[kt][8 + i] = r1[i]; }
    }

    // x values for this lane's D rows (p = vv + half*8)
    float xv[8];
    {
      const float* xp = &sX[w][xbase + half * 8];
#pragma unroll
      for (int i = 0; i < 8; ++i) xv[i] = xp[i];
    }

    // -- phase D/E: backward g2 @ W2^T, chain to df partials --
    float dfp[8];
#pragma unroll
    for (int vv = 0; vv < 8; ++vv) dfp[vv] = 0.0f;

#pragma unroll
    for (int nt = 0; nt < 4; ++nt) {
      v8f acc2;
#pragma unroll
      for (int vv = 0; vv < 8; ++vv) acc2[vv] = 0.0f;
#pragma unroll
      for (int kt = 0; kt < 2; ++kt) {
        // W2^T fragment: B'[k][j] = W2[j][k], j = nt*16+n fixed -> contiguous row of W2
        const _Float16* wr = &sW2h[(nt * 16 + n) * 64 + kt * 32 + half * 8];
        h8 q0 = *(const h8*)(wr);
        h8 q1 = *(const h8*)(wr + 16);
        v16h bt;
#pragma unroll
        for (int i = 0; i < 8; ++i) { bt[i] = q0[i]; bt[8 + i] = q1[i]; }
        acc2 = __builtin_amdgcn_wmma_f32_16x16x32_f16(false, ap[kt], false, bt,
                                                      (short)0, acc2, false, false);
      }
      // df partial: g1 * silu'(z1) * W1[k], k = nt*16 + n, row p = vv + half*8
#pragma unroll
      for (int vv = 0; vv < 8; ++vv) {
        float z1 = fmaf(xv[vv], w1c[nt], b1c[nt]);
        float s  = sigmoid_(z1);
        float sp = s * (1.0f + z1 * (1.0f - s));
        dfp[vv]  = fmaf(acc2[vv], sp * w1c[nt], dfp[vv]);
      }
    }

    // -- cross-lane reduction over the 16-lane half (K dimension) --
#pragma unroll
    for (int vv = 0; vv < 8; ++vv) {
      float fs = fp[vv], ds = dfp[vv];
#pragma unroll
      for (int m = 1; m < 16; m <<= 1) {
        fs += __shfl_xor(fs, m, 32);
        ds += __shfl_xor(ds, m, 32);
      }
      fp[vv] = fs; dfp[vv] = ds;
    }
    if (n == 0) {
#pragma unroll
      for (int vv = 0; vv < 8; ++vv) {
        const int slot = xbase + half * 8 + vv;
        sFD[w][0][slot] = fp[vv] + b3v;
        sFD[w][1][slot] = dfp[vv];
      }
    }
    __syncthreads();
  }

  // ---- H per edge ----
  const float fL  = sFD[w][0][l];
  const float fR  = sFD[w][0][32 + l];
  const float dfL = sFD[w][1][l];
  const float dfR = sFD[w][1][32 + l];
  const float rho = fmaxf(fabsf(dfL), fabsf(dfR));
  Hout[(size_t)row * NPOS + e] = 0.5f * (fR + fL - rho * (uRv - uLv));
}

// out = ca*u0 + cb*v - cf*(H[i+1]-H[i]),  cf = cc*DT/DX, periodic in i.
__global__ __launch_bounds__(256) void kt_update_kernel(
    const float* __restrict__ u0, const float* __restrict__ vin,
    const float* __restrict__ H, float ca, float cb, float cf,
    float* __restrict__ out)
{
  const int idx = blockIdx.x * 256 + threadIdx.x;
  const int rb  = idx & ~(NPOS - 1);
  const int i   = idx & (NPOS - 1);
  const float h0 = H[idx];
  const float h1 = H[rb | ((i + 1) & (NPOS - 1))];
  out[idx] = ca * u0[idx] + cb * vin[idx] - cf * (h1 - h0);
}

extern "C" void kernel_launch(void* const* d_in, const int* in_sizes, int n_in,
                              void* d_out, int out_size, void* d_ws, size_t ws_size,
                              hipStream_t stream) {
  (void)in_sizes; (void)n_in; (void)out_size; (void)ws_size;
  const float* u  = (const float*)d_in[0];
  const float* W1 = (const float*)d_in[1];
  const float* B1 = (const float*)d_in[2];
  const float* W2 = (const float*)d_in[3];
  const float* B2 = (const float*)d_in[4];
  const float* W3 = (const float*)d_in[5];
  const float* B3 = (const float*)d_in[6];
  float* out = (float*)d_out;

  float* Hbuf = (float*)d_ws;                       // 128*8192 f32 = 4 MiB
  float* ubuf = Hbuf + (size_t)NROWS * NPOS;        // RK intermediate state

  const float r = 0.0004f / 0.001f;                 // DT/DX
  dim3 gH(NPOS / (WAVES * 32), NROWS), bH(256);
  dim3 gU((NROWS * NPOS) / 256), bU(256);

  // stage 1: u1 = u + DT*rhs(u)
  kt_flux_kernel<<<gH, bH, 0, stream>>>(u, W1, B1, W2, B2, W3, B3, Hbuf);
  kt_update_kernel<<<gU, bU, 0, stream>>>(u, u, Hbuf, 1.0f, 0.0f, 1.0f * r, ubuf);
  // stage 2: u2 = 0.75u + 0.25u1 + 0.25*DT*rhs(u1)   (in-place into ubuf)
  kt_flux_kernel<<<gH, bH, 0, stream>>>(ubuf, W1, B1, W2, B2, W3, B3, Hbuf);
  kt_update_kernel<<<gU, bU, 0, stream>>>(u, ubuf, Hbuf, 0.75f, 0.25f, 0.25f * r, ubuf);
  // stage 3: out = u/3 + 2/3 u2 + 2/3 DT*rhs(u2)
  kt_flux_kernel<<<gH, bH, 0, stream>>>(ubuf, W1, B1, W2, B2, W3, B3, Hbuf);
  kt_update_kernel<<<gU, bU, 0, stream>>>(u, ubuf, Hbuf, 1.0f / 3.0f, 2.0f / 3.0f,
                                          (2.0f / 3.0f) * r, out);
}
